// BiGRU_CONCEPT_CNN_82368882803024
// MI455X (gfx1250) — matrix operand order repaired and checked by reading, not compiled, not run on hardware
//
#include <hip/hip_runtime.h>
#include <hip/hip_bf16.h>

// ---------------------------------------------------------------------------
// BiGRU + concept-attention + CNN classifier for MI455X (gfx1250, wave32).
// All large GEMMs use V_WMMA_F32_16X16X32_BF16 (f32 accumulate).
// B matrices are pre-padded (Kpad mult 32, Npad mult 128, zero-filled) so the
// GEMM main loop is completely branch-free; LDS is double-buffered.
// ---------------------------------------------------------------------------

typedef __attribute__((ext_vector_type(16))) __bf16          v16bf;
typedef __attribute__((ext_vector_type(8)))  unsigned        v8u;   // 32B raw frag
typedef __attribute__((ext_vector_type(8)))  float           v8f;

#define Bsz 64
#define Tsz 128
#define Dsz 300
#define Hsz 256
#define Ksz 16

static __device__ __forceinline__ unsigned short f2bf_bits(float f) {
  unsigned u = __builtin_bit_cast(unsigned, f);
  unsigned r = u + 0x7FFFu + ((u >> 16) & 1u);   // round-to-nearest-even
  return (unsigned short)(r >> 16);
}

// ---------------------------------------------------------------------------
// Prep kernels
// ---------------------------------------------------------------------------
__global__ void k_embed(const int* __restrict__ inp, const float* __restrict__ emb,
                        unsigned short* __restrict__ xb) {
  const size_t total = (size_t)Bsz * Tsz * Dsz;
  for (size_t i = (size_t)blockIdx.x * blockDim.x + threadIdx.x; i < total;
       i += (size_t)gridDim.x * blockDim.x) {
    size_t row = i / Dsz, d = i % Dsz;
    int v = inp[row];
    xb[i] = f2bf_bits(emb[(size_t)v * Dsz + d]);
  }
}

// src [Nn, Kk] f32 -> dst [Kpad, Npad] bf16, transposed + zero-padded.
__global__ void k_transpose_pad(const float* __restrict__ src, unsigned short* __restrict__ dst,
                                int Nn, int Kk, int Npad, int Kpad) {
  const size_t total = (size_t)Npad * Kpad;
  for (size_t i = (size_t)blockIdx.x * blockDim.x + threadIdx.x; i < total;
       i += (size_t)gridDim.x * blockDim.x) {
    size_t k = i / Npad, n = i % Npad;
    unsigned short v = 0;
    if ((int)n < Nn && (int)k < Kk) v = f2bf_bits(src[n * (size_t)Kk + k]);
    dst[i] = v;
  }
}

// Pack Wh [768,256] into WMMA-B-fragment-native layout:
// dst[((kt*48+nt)*32+lane)*16 + e] so each lane's fragment is one 32B load.
// Contiguous K order used by this code (A matches): lanes 0-15 hold K=0..15 of tile, lanes 16-31 K=16..31,
// col = lane&15; VGPR v holds K pair (2v, 2v+1) in [15:0],[31:16].
__global__ void k_pack_wh(const float* __restrict__ Wh, unsigned short* __restrict__ dst) {
  const int total = 8 * 48 * 32 * 16;   // ktiles * ntiles * lanes * elems
  int i = blockIdx.x * blockDim.x + threadIdx.x;
  if (i >= total) return;
  int e    = i & 15;
  int lane = (i >> 4) & 31;
  int nt   = (i >> 9) % 48;
  int kt   = (i >> 9) / 48;
  int v = e >> 1, odd = e & 1, half = lane >> 4, n15 = lane & 15;
  int kb = 2 * v + (half ? 16 : 0) + odd;
  int Kc = kt * 32 + kb;          // 0..255
  int Nc = nt * 16 + n15;         // 0..767
  dst[i] = f2bf_bits(Wh[(size_t)Nc * Hsz + Kc]);   // B[k][n] = Wh[n][k]
}

// ---------------------------------------------------------------------------
// Branch-free WMMA GEMM: C[M,N](f32) = A[M,Kpad](bf16) * B[Kpad,Npad](bf16) + bias
// Requirements: M % 64 == 0, Kpad % 32 == 0 and >= 64, Npad % 128 == 0,
// B zero-padded, A rows readable up to Kpad elements (slack allocation).
// Block: 256 threads = 8 waves; wave tile 32x32; workgroup tile 64x128.
// As: row-major [64][AST]; Bs: K-pair interleaved u32 [16 pairs][128 cols];
// LDS double-buffered: stage tile k+1 while computing tile k (1 barrier/iter).
// ---------------------------------------------------------------------------
#define AST 36

template <bool RELU, bool HASCO>
__global__ __launch_bounds__(256) void k_gemm(
    const unsigned short* __restrict__ A, int lda,
    const unsigned short* __restrict__ B, int ldb,   // ldb == Npad
    const float* __restrict__ bias,
    float* __restrict__ C, int ldc,
    unsigned short* __restrict__ Co, int ldco,
    int M, int N, int Kpad) {
  __shared__ unsigned short As[2][64 * AST];
  __shared__ unsigned       Bs32[2][16 * 128];
  const int tid  = threadIdx.x;
  const int lane = tid & 31;
  const int w    = tid >> 5;
  const int wm   = w >> 2, wn = w & 3;
  const int mBase = blockIdx.y * 64;
  const int nBase = blockIdx.x * 128;
  const int half = lane >> 4;
  const int l15  = lane & 15;

  v8f acc[2][2];
#pragma unroll
  for (int i = 0; i < 2; ++i)
#pragma unroll
    for (int j = 0; j < 2; ++j)
#pragma unroll
      for (int r = 0; r < 8; ++r) acc[i][j][r] = 0.f;

  // staging index precompute
  const int ar  = tid >> 2;            // A row 0..63
  const int akk = (tid & 3) * 8;       // A k-chunk
  const int bp  = tid >> 4;            // B K-pair 0..15
  const int bn  = (tid & 15) * 8;      // B n-chunk 0..120

  const unsigned short* aSrc = A + (size_t)(mBase + ar) * lda + akk;
  const unsigned short* bSrc = B + (size_t)(2 * bp) * ldb + nBase + bn;

  auto stage = [&](int buf, int k0) {
    {  // A tile 64x32
      const unsigned short* s = aSrc + k0;
      uint2 v0 = *(const uint2*)(s);
      uint2 v1 = *(const uint2*)(s + 4);
      *(uint2*)&As[buf][ar * AST + akk]     = v0;
      *(uint2*)&As[buf][ar * AST + akk + 4] = v1;
    }
    {  // B tile 32x128, pair-interleave, contiguous u32 stores
      const unsigned short* s0 = bSrc + (size_t)k0 * ldb;
      const unsigned short* s1 = s0 + ldb;
      uint2 r00 = *(const uint2*)(s0);
      uint2 r01 = *(const uint2*)(s0 + 4);
      uint2 r10 = *(const uint2*)(s1);
      uint2 r11 = *(const uint2*)(s1 + 4);
      unsigned short e0[8], e1[8];
      *(uint2*)&e0[0] = r00; *(uint2*)&e0[4] = r01;
      *(uint2*)&e1[0] = r10; *(uint2*)&e1[4] = r11;
#pragma unroll
      for (int j = 0; j < 8; ++j)
        Bs32[buf][bp * 128 + bn + j] = (unsigned)e0[j] | ((unsigned)e1[j] << 16);
    }
  };

  auto compute = [&](int buf) {
    v8u a[2], b[2];
#pragma unroll
    for (int i = 0; i < 2; ++i) {  // A fragments: 8x u32 (pairs contiguous)
      int mrow = wm * 32 + i * 16 + l15;
#pragma unroll
      for (int v = 0; v < 8; ++v) {
        int kb = 2 * v + (half ? 16 : 0);
        a[i][v] = *(const unsigned*)&As[buf][mrow * AST + kb];
      }
    }
#pragma unroll
    for (int i = 0; i < 2; ++i) {  // B fragments: 8x u32 from pair layout
      int ncol = wn * 32 + i * 16 + l15;
#pragma unroll
      for (int v = 0; v < 8; ++v) {
        int p = v + (half ? 8 : 0);
        b[i][v] = Bs32[buf][p * 128 + ncol];
      }
    }
#pragma unroll
    for (int i = 0; i < 2; ++i)
#pragma unroll
      for (int j = 0; j < 2; ++j)
        acc[i][j] = __builtin_amdgcn_wmma_f32_16x16x32_bf16(
            false, __builtin_bit_cast(v16bf, a[i]),
            false, __builtin_bit_cast(v16bf, b[j]),
            (short)0, acc[i][j], false, false);
  };

  // double-buffered pipeline
  stage(0, 0);
  __syncthreads();
  int buf = 0;
#pragma unroll 1
  for (int k0 = 32; k0 < Kpad; k0 += 32) {
    stage(buf ^ 1, k0);   // prefetch next tile while computing current
    compute(buf);
    __syncthreads();
    buf ^= 1;
  }
  compute(buf);

  // epilogue: bias (+ReLU) (+bf16 copy-out)
#pragma unroll
  for (int j = 0; j < 2; ++j) {
    int n = nBase + wn * 32 + j * 16 + l15;
    if (n < N) {
      float bv = bias[n];
#pragma unroll
      for (int i = 0; i < 2; ++i) {
        int mrow = mBase + wm * 32 + i * 16 + (half ? 8 : 0);
        float* cp = C + (size_t)mrow * ldc + n;
#pragma unroll
        for (int r = 0; r < 8; ++r) {
          float v = acc[i][j][r] + bv;
          if (RELU) v = fmaxf(v, 0.f);
          cp[(size_t)r * ldc] = v;
          if (HASCO) Co[(size_t)(mrow + r) * ldco + n] = f2bf_bits(v);
        }
      }
    }
  }
}

// ---------------------------------------------------------------------------
// GRU recurrence: 8 blocks = 2 dirs x 4 batch-groups of 16 rows.
// Per step: gh[16,768] = h_bf16[16,256] @ WhT (WMMA), then gate math.
// h kept in registers (16 f32 per thread); gh staged in 48KB LDS.
// The kk (K-tile) loop is NOT unrolled to bound live registers:
// 48 acc + 8 a + 48 in-flight B frags ~ 120 VGPRs (no spills).
// ---------------------------------------------------------------------------
__global__ __launch_bounds__(256) void k_gru(
    const unsigned short* __restrict__ WhPf, const unsigned short* __restrict__ WhPb,
    const float* __restrict__ xg_f, const float* __restrict__ xg_b,
    const float* __restrict__ bh_f, const float* __restrict__ bh_b,
    unsigned short* __restrict__ outb) {
  extern __shared__ char smem[];
  float* gh          = (float*)smem;                         // 16*768 f32
  unsigned short* hb = (unsigned short*)(gh + 16 * 768);     // 16*258 bf16 bits

  const int dir = blockIdx.x >> 2;
  const int bg  = blockIdx.x & 3;
  const float* xg = dir ? xg_b : xg_f;
  const float* bh = dir ? bh_b : bh_f;
  const unsigned short* Wp = dir ? WhPb : WhPf;

  const int tid = threadIdx.x, lane = tid & 31, w = tid >> 5;
  const int half = lane >> 4, l15 = lane & 15;

  float hreg[16];
#pragma unroll
  for (int i = 0; i < 16; ++i) hreg[i] = 0.f;
  for (int i = tid; i < 16 * 258; i += 256) hb[i] = 0;
  __syncthreads();

#pragma unroll 1
  for (int s = 0; s < Tsz; ++s) {
    const int t = dir ? (Tsz - 1 - s) : s;

    // prefetch this step's gate rows while the WMMA phase runs
    {
      const float* pf = xg + ((size_t)(bg * 16 + (tid >> 4)) * Tsz + t) * 768
                        + (tid & 15) * 48;
      __builtin_prefetch(pf, 0, 1);
      __builtin_prefetch(pf + 32, 0, 1);
    }

    // -------- phase 1: gh = h @ Wh^T (48 N-tiles, 6 per wave) --------
    v8f acc[6];
#pragma unroll
    for (int j = 0; j < 6; ++j)
#pragma unroll
      for (int r = 0; r < 8; ++r) acc[j][r] = 0.f;

#pragma unroll 1
    for (int kk = 0; kk < 8; ++kk) {
      // A fragment for this K-tile (8x u32 from LDS; pairs contiguous)
      v8u a;
#pragma unroll
      for (int v = 0; v < 8; ++v) {
        int kb = kk * 32 + 2 * v + (half ? 16 : 0);
        a[v] = *(const unsigned*)&hb[l15 * 258 + kb];
      }
      const unsigned short* wrow =
          Wp + (((size_t)(kk * 48 + w * 6)) * 32 + lane) * 16;
#pragma unroll
      for (int j = 0; j < 6; ++j) {
        v8u braw = *(const v8u*)(wrow + (size_t)j * (32 * 16));
        acc[j] = __builtin_amdgcn_wmma_f32_16x16x32_bf16(
            false, __builtin_bit_cast(v16bf, a),
            false, __builtin_bit_cast(v16bf, braw),
            (short)0, acc[j], false, false);
      }
    }
#pragma unroll
    for (int j = 0; j < 6; ++j) {
      int n = (w * 6 + j) * 16 + l15;
#pragma unroll
      for (int r = 0; r < 8; ++r)
        gh[(r + (half ? 8 : 0)) * 768 + n] = acc[j][r];
    }
    __syncthreads();

    // -------- phase 2: gate math, h update --------
#pragma unroll
    for (int i = 0; i < 16; ++i) {
      int idx = tid + i * 256;
      int m = idx >> 8, jj = idx & 255;
      int gm = bg * 16 + m;
      const float* xr = xg + ((size_t)gm * Tsz + t) * 768;
      float hr = gh[m * 768 + jj]       + bh[jj];
      float hz = gh[m * 768 + 256 + jj] + bh[256 + jj];
      float hn = gh[m * 768 + 512 + jj] + bh[512 + jj];
      float r  = 1.f / (1.f + __expf(-(xr[jj] + hr)));
      float z  = 1.f / (1.f + __expf(-(xr[256 + jj] + hz)));
      float nn = tanhf(xr[512 + jj] + r * hn);
      float h2 = (1.f - z) * nn + z * hreg[i];
      hreg[i] = h2;
      unsigned short hbits = f2bf_bits(h2);
      hb[m * 258 + jj] = hbits;
      outb[((size_t)gm * Tsz + t) * 512 + dir * 256 + jj] = hbits;
    }
    __syncthreads();
  }
}

// ---------------------------------------------------------------------------
// Concept attention: per token gather concept_table[v], score vs ctx,
// masked softmax over K=16, weighted sum -> feat[:,300:600] (bf16).
// ---------------------------------------------------------------------------
__global__ __launch_bounds__(128) void k_attn(
    const int* __restrict__ inp, const float* __restrict__ ctx,
    const float* __restrict__ ctab, const unsigned char* __restrict__ cmask,
    unsigned short* __restrict__ feat) {
  __shared__ float ctxs[Dsz];
  __shared__ float red[128];
  __shared__ float sc[Ksz];
  const int row = blockIdx.x;
  const int tid = threadIdx.x;
  const int v = inp[row];
  const float* crow = ctx + (size_t)row * Dsz;
  const float* tab  = ctab + (size_t)v * Ksz * Dsz;

  for (int d = tid; d < Dsz; d += 128) ctxs[d] = crow[d];
  __syncthreads();

  for (int k = 0; k < Ksz; ++k) {
    float p = 0.f;
    for (int d = tid; d < Dsz; d += 128) p += tab[k * Dsz + d] * ctxs[d];
    red[tid] = p;
    __syncthreads();
    for (int off = 64; off > 0; off >>= 1) {
      if (tid < off) red[tid] += red[tid + off];
      __syncthreads();
    }
    if (tid == 0) sc[k] = cmask[(size_t)v * Ksz + k] ? red[0] : -1e30f;
    __syncthreads();
  }
  if (tid == 0) {
    float mx = sc[0];
    for (int k = 1; k < Ksz; ++k) mx = sc[k] > mx ? sc[k] : mx;
    float sum = 0.f;
    for (int k = 0; k < Ksz; ++k) { float e = __expf(sc[k] - mx); sc[k] = e; sum += e; }
    float inv = 1.f / sum;
    for (int k = 0; k < Ksz; ++k) sc[k] *= inv;
  }
  __syncthreads();
  for (int d = tid; d < Dsz; d += 128) {
    float acc = 0.f;
#pragma unroll
    for (int k = 0; k < Ksz; ++k) acc += sc[k] * tab[k * Dsz + d];
    feat[(size_t)row * 600 + Dsz + d] = f2bf_bits(acc);
  }
}

// ---------------------------------------------------------------------------
// Max-pool over conv positions + tiny MLP head with softmax
// ---------------------------------------------------------------------------
__global__ void k_pool(const float* __restrict__ cf0, const float* __restrict__ cf1,
                       const float* __restrict__ cf2, float* __restrict__ pooled) {
  int b = blockIdx.x, c = blockIdx.y, f = threadIdx.x;
  if (f >= 100) return;
  const float* cf = (c == 0) ? cf0 : (c == 1) ? cf1 : cf2;
  int L = Tsz - ((c == 0) ? 3 : (c == 1) ? 4 : 5) + 1;
  float m = -1e30f;
  for (int l = 0; l < L; ++l) {
    float v = cf[(size_t)(b * Tsz + l) * 100 + f];
    m = v > m ? v : m;
  }
  pooled[b * 300 + c * 100 + f] = m;
}

__global__ __launch_bounds__(128) void k_head(
    const float* __restrict__ pooled, const float* __restrict__ fc1W,
    const float* __restrict__ fc1b, const float* __restrict__ fc2W,
    const float* __restrict__ fc2b, float* __restrict__ outp) {
  __shared__ float h1[100];
  __shared__ float o[5];
  int b = blockIdx.x, tid = threadIdx.x;
  if (tid < 100) {
    float s = fc1b[tid];
    for (int k = 0; k < 300; ++k) s += pooled[b * 300 + k] * fc1W[tid * 300 + k];
    h1[tid] = s;
  }
  __syncthreads();
  if (tid < 5) {
    float s = fc2b[tid];
    for (int k = 0; k < 100; ++k) s += h1[k] * fc2W[tid * 100 + k];
    o[tid] = s;
  }
  __syncthreads();
  if (tid == 0) {
    float mx = o[0];
    for (int i = 1; i < 5; ++i) mx = o[i] > mx ? o[i] : mx;
    float e[5], sum = 0.f;
    for (int i = 0; i < 5; ++i) { e[i] = __expf(o[i] - mx); sum += e[i]; }
    float inv = 1.f / sum;
    for (int i = 0; i < 5; ++i) outp[b * 5 + i] = e[i] * inv;
  }
}

// ---------------------------------------------------------------------------
// Host orchestration
// ---------------------------------------------------------------------------
extern "C" void kernel_launch(void* const* d_in, const int* in_sizes, int n_in,
                              void* d_out, int out_size, void* d_ws, size_t ws_size,
                              hipStream_t stream) {
  (void)in_sizes; (void)n_in; (void)out_size; (void)ws_size;
  const int*   inp     = (const int*)  d_in[0];
  const float* emb     = (const float*)d_in[1];
  const float* Wx_f    = (const float*)d_in[2];
  const float* Wh_f    = (const float*)d_in[3];
  const float* bx_f    = (const float*)d_in[4];
  const float* bh_f    = (const float*)d_in[5];
  const float* Wx_b    = (const float*)d_in[6];
  const float* Wh_b    = (const float*)d_in[7];
  const float* bx_b    = (const float*)d_in[8];
  const float* bh_b    = (const float*)d_in[9];
  const float* fc1c_W  = (const float*)d_in[10];
  const float* fc1c_b  = (const float*)d_in[11];
  const float* ctab    = (const float*)d_in[12];
  const unsigned char* cmask = (const unsigned char*)d_in[13];
  const float* conv_W0 = (const float*)d_in[14];
  const float* conv_b0 = (const float*)d_in[15];
  const float* conv_W1 = (const float*)d_in[16];
  const float* conv_b1 = (const float*)d_in[17];
  const float* conv_W2 = (const float*)d_in[18];
  const float* conv_b2 = (const float*)d_in[19];
  const float* fc1_W   = (const float*)d_in[20];
  const float* fc1_b   = (const float*)d_in[21];
  const float* fc2_W   = (const float*)d_in[22];
  const float* fc2_b   = (const float*)d_in[23];

  // workspace carving (256B aligned)
  size_t off = 0;
  auto carve = [&](size_t bytes) -> void* {
    void* p = (char*)d_ws + off;
    off = (off + bytes + 255) & ~(size_t)255;
    return p;
  };
  const size_t BT = (size_t)Bsz * Tsz;  // 8192
  // A-side activations get slack so unguarded reads up to Kpad stay in-bounds.
  unsigned short* xb    = (unsigned short*)carve((BT * Dsz + 128) * 2);
  unsigned short* WxTf  = (unsigned short*)carve((size_t)320 * 768 * 2);
  unsigned short* WxTb  = (unsigned short*)carve((size_t)320 * 768 * 2);
  unsigned short* WhPf  = (unsigned short*)carve((size_t)8 * 48 * 32 * 16 * 2);
  unsigned short* WhPb  = (unsigned short*)carve((size_t)8 * 48 * 32 * 16 * 2);
  unsigned short* fc1cT = (unsigned short*)carve((size_t)512 * 384 * 2);
  unsigned short* WcT0  = (unsigned short*)carve((size_t)1824 * 128 * 2);
  unsigned short* WcT1  = (unsigned short*)carve((size_t)2400 * 128 * 2);
  unsigned short* WcT2  = (unsigned short*)carve((size_t)3008 * 128 * 2);
  float*          xgf   = (float*)carve(BT * 768 * 4);
  float*          xgb   = (float*)carve(BT * 768 * 4);
  unsigned short* outb  = (unsigned short*)carve(BT * 512 * 2);
  float*          ctx   = (float*)carve(BT * Dsz * 4);
  unsigned short* feat  = (unsigned short*)carve((BT + 16) * 600 * 2);  // +pad rows for conv overread
  float*          cf0   = (float*)carve(BT * 100 * 4);
  float*          cf1   = (float*)carve(BT * 100 * 4);
  float*          cf2   = (float*)carve(BT * 100 * 4);
  float*          pooled= (float*)carve((size_t)Bsz * 300 * 4);

  // 1) embedding gather -> bf16
  k_embed<<<2400, 256, 0, stream>>>(inp, emb, xb);

  // 2) weight conversions (transposed, zero-padded to Kpad x Npad)
  k_transpose_pad<<<512, 256, 0, stream>>>(Wx_f,    WxTf,  768, Dsz,  768, 320);
  k_transpose_pad<<<512, 256, 0, stream>>>(Wx_b,    WxTb,  768, Dsz,  768, 320);
  k_transpose_pad<<<512, 256, 0, stream>>>(fc1c_W,  fc1cT, Dsz, 512,  384, 512);
  k_transpose_pad<<<512, 256, 0, stream>>>(conv_W0, WcT0,  100, 1800, 128, 1824);
  k_transpose_pad<<<512, 256, 0, stream>>>(conv_W1, WcT1,  100, 2400, 128, 2400);
  k_transpose_pad<<<512, 256, 0, stream>>>(conv_W2, WcT2,  100, 3000, 128, 3008);
  k_pack_wh<<<768, 256, 0, stream>>>(Wh_f, WhPf);
  k_pack_wh<<<768, 256, 0, stream>>>(Wh_b, WhPb);

  // 3) input-gate GEMMs: xg = x @ Wx^T + bx   [8192 x 768], Kpad=320
  {
    dim3 g(768 / 128, (int)(BT / 64));
    k_gemm<false, false><<<g, 256, 0, stream>>>(xb, Dsz, WxTf, 768, bx_f, xgf, 768,
                                                nullptr, 0, (int)BT, 768, 320);
    k_gemm<false, false><<<g, 256, 0, stream>>>(xb, Dsz, WxTb, 768, bx_b, xgb, 768,
                                                nullptr, 0, (int)BT, 768, 320);
  }

  // 4) GRU recurrence (both directions) -> out bf16 [8192 x 512]
  {
    size_t lds = (size_t)16 * 768 * 4 + (size_t)16 * 258 * 2;  // ~57.4KB
    k_gru<<<8, 256, lds, stream>>>(WhPf, WhPb, xgf, xgb, bh_f, bh_b, outb);
  }

  // 5) ctx = out @ fc1c_W^T + b  -> ctx f32 + feat[:,0:300] bf16 (Npad=384)
  {
    dim3 g(384 / 128, (int)(BT / 64));
    k_gemm<false, true><<<g, 256, 0, stream>>>(outb, 512, fc1cT, 384, fc1c_b, ctx, Dsz,
                                               feat, 600, (int)BT, Dsz, 512);
  }

  // 6) concept attention -> feat[:,300:600]
  k_attn<<<(int)BT, 128, 0, stream>>>(inp, ctx, ctab, cmask, feat);

  // 7) conv GEMMs over token windows (contiguous slices of feat) + ReLU
  {
    dim3 g(1, (int)(BT / 64));
    k_gemm<true, false><<<g, 256, 0, stream>>>(feat, 600, WcT0, 128, conv_b0, cf0, 100,
                                               nullptr, 0, (int)BT, 100, 1824);
    k_gemm<true, false><<<g, 256, 0, stream>>>(feat, 600, WcT1, 128, conv_b1, cf1, 100,
                                               nullptr, 0, (int)BT, 100, 2400);
    k_gemm<true, false><<<g, 256, 0, stream>>>(feat, 600, WcT2, 128, conv_b2, cf2, 100,
                                               nullptr, 0, (int)BT, 100, 3008);
  }

  // 8) max-pool + head
  {
    dim3 g(Bsz, 3);
    k_pool<<<g, 128, 0, stream>>>(cf0, cf1, cf2, pooled);
  }
  k_head<<<Bsz, 128, 0, stream>>>(pooled, fc1_W, fc1_b, fc2_W, fc2_b, (float*)d_out);
}
